// Attend_21328807592661
// MI455X (gfx1250) — matrix operand order repaired and checked by reading, not compiled, not run on hardware
//
#include <hip/hip_runtime.h>
#include <hip/hip_bf16.h>

#define D_IN  128
#define D_HID 256
#define NB    4
#define SEQ   4096

typedef __attribute__((ext_vector_type(16))) __bf16 v16bf;
typedef __attribute__((ext_vector_type(8)))  __bf16 v8bf;
typedef __attribute__((ext_vector_type(8)))  float  v8f;

static __device__ __forceinline__ v16bf bfcat(v8bf a, v8bf b) {
  return __builtin_shufflevector(a, b, 0,1,2,3,4,5,6,7,8,9,10,11,12,13,14,15);
}

static __device__ __forceinline__ v8f fzero() {
  v8f z;
#pragma unroll
  for (int i = 0; i < 8; ++i) z[i] = 0.f;
  return z;
}

static __device__ __forceinline__ v8f wmma_bf16(v16bf a, v16bf b, v8f c) {
  return __builtin_amdgcn_wmma_f32_16x16x32_bf16(false, a, false, b, (short)0, c, false, false);
}

// A-fragment (16x32 bf16, MxK) from row-major bf16 [row][k], row stride ld.
// Lane l: row = l&15, K chunks (l>>4)*8+{0..7} and 16+(l>>4)*8+{0..7}.
static __device__ __forceinline__ v16bf load_a_bf16(const __bf16* base, int ld, int kb, int lane) {
  int m = lane & 15, kq = (lane >> 4) * 8;
  const __bf16* p = base + (size_t)m * ld + kb + kq;
  return bfcat(*(const v8bf*)p, *(const v8bf*)(p + 16));
}

// A-fragment from row-major f32 source (converted to bf16 on the fly).
static __device__ __forceinline__ v16bf load_a_f32(const float* base, int ld, int kb, int lane) {
  int m = lane & 15, kq = (lane >> 4) * 8;
  const float* p0 = base + (size_t)m * ld + kb + kq;
  v8f f0 = *(const v8f*)p0;
  v8f f1 = *(const v8f*)(p0 + 16);
  v16bf a;
#pragma unroll
  for (int i = 0; i < 8; ++i) { a[i] = (__bf16)f0[i]; a[8 + i] = (__bf16)f1[i]; }
  return a;
}

// B-fragment (32x16 bf16, KxN) from "K-contiguous" storage baseT[n][k], row stride ld.
// Lane l: column n = l&15, K chunks (l>>4)*8+{0..7} and 16+(l>>4)*8+{0..7}.
static __device__ __forceinline__ v16bf load_b_bf16(const __bf16* baseT, int ld, int kb, int lane) {
  int n = lane & 15, ko = (lane >> 4) * 8;
  const __bf16* p = baseT + (size_t)n * ld + kb + ko;
  return bfcat(*(const v8bf*)p, *(const v8bf*)(p + 16));
}

// ---------------------------------------------------------------------------
// Prep: bf16-transpose weights: W1T[n][k] = W1[k][n], W2T[n][k] = W2[k][n]
// ---------------------------------------------------------------------------
__global__ __launch_bounds__(256) void prep_weights_kernel(
    const float* __restrict__ W1, const float* __restrict__ W2,
    __bf16* __restrict__ W1T, __bf16* __restrict__ W2T) {
  int t = blockIdx.x * blockDim.x + threadIdx.x;
  int stride = gridDim.x * blockDim.x;
  for (int i = t; i < D_HID * D_IN; i += stride) {
    int n = i / D_IN, k = i - n * D_IN;
    W1T[i] = (__bf16)W1[k * D_HID + n];
  }
  for (int i = t; i < D_HID * D_HID; i += stride) {
    int n = i / D_HID, k = i - n * D_HID;
    W2T[i] = (__bf16)W2[k * D_HID + n];
  }
}

// ---------------------------------------------------------------------------
// Prep: Xt[b][d][n] = (bf16) X[b][n][d]   (so V-columns are K-contiguous)
// ---------------------------------------------------------------------------
__global__ __launch_bounds__(256) void transpose_kernel(
    const float* __restrict__ X, __bf16* __restrict__ Xt) {
  size_t total = (size_t)NB * SEQ * D_IN;
  for (size_t o = (size_t)blockIdx.x * blockDim.x + threadIdx.x; o < total;
       o += (size_t)gridDim.x * blockDim.x) {
    size_t b = o / ((size_t)D_IN * SEQ);
    size_t r = o - b * (size_t)D_IN * SEQ;
    size_t d = r / SEQ, n = r - d * SEQ;
    Xt[o] = (__bf16)X[(b * SEQ + n) * D_IN + d];
  }
}

// ---------------------------------------------------------------------------
// MLP: F[row][0:256] = relu(relu(X[row]@W1+b1)@W2+b2), bf16 out.
// One wave per 16-row tile; hidden activations round-trip through LDS so the
// second GEMM can re-load them in A-fragment layout.
// ---------------------------------------------------------------------------
__global__ __launch_bounds__(128) void mlp_kernel(
    const float* __restrict__ X,
    const __bf16* __restrict__ W1T, const float* __restrict__ b1,
    const __bf16* __restrict__ W2T, const float* __restrict__ b2,
    __bf16* __restrict__ F) {
  __shared__ __align__(32) __bf16 hbuf[4][16][D_HID];
  int lane = threadIdx.x & 31;
  int w = threadIdx.x >> 5;
  int row0 = (blockIdx.x * 4 + w) * 16;
  const float* x = X + (size_t)row0 * D_IN;
  int nl = lane & 15, hh = lane >> 4;

  // layer 1: 16x256 = 16 n-tiles, K=128
#pragma unroll 4
  for (int nt = 0; nt < D_HID / 16; ++nt) {
    int n0 = nt * 16;
    v8f acc = fzero();
#pragma unroll
    for (int kb = 0; kb < D_IN; kb += 32) {
      v16bf a = load_a_f32(x, D_IN, kb, lane);
      v16bf bf = load_b_bf16(W1T + (size_t)n0 * D_IN, D_IN, kb, lane);
      acc = wmma_bf16(a, bf, acc);
    }
    float bias = b1[n0 + nl];
#pragma unroll
    for (int v = 0; v < 8; ++v) {
      float t = fmaxf(acc[v] + bias, 0.f);
      hbuf[w][v + hh * 8][n0 + nl] = (__bf16)t;
    }
  }

  // layer 2: 16x256, K=256 (A from LDS; same-wave LDS is in-order)
  const __bf16* hp = &hbuf[w][0][0];
#pragma unroll 4
  for (int nt = 0; nt < D_HID / 16; ++nt) {
    int n0 = nt * 16;
    v8f acc = fzero();
#pragma unroll
    for (int kb = 0; kb < D_HID; kb += 32) {
      v16bf a = load_a_bf16(hp, D_HID, kb, lane);
      v16bf bf = load_b_bf16(W2T + (size_t)n0 * D_HID, D_HID, kb, lane);
      acc = wmma_bf16(a, bf, acc);
    }
    float bias = b2[n0 + nl];
#pragma unroll
    for (int v = 0; v < 8; ++v) {
      float t = fmaxf(acc[v] + bias, 0.f);
      F[(size_t)(row0 + v + hh * 8) * D_HID + n0 + nl] = (__bf16)t;
    }
  }
}

// ---------------------------------------------------------------------------
// Fused flash kernel:
//   out[m][d] = sum_n softmax_row(FQ·FKᵀ)[m][n] * V[n][d]
// One wave per 16 rows; streams all 4096 columns with online softmax:
//  - e tile (16x32) via WMMA (FQ A-frags register-resident, FK from L2)
//  - row max made lane-uniform via 16-lane shfl_xor reduce
//  - output accumulators rescaled by exp(old_max - new_max)
//  - P converted C-layout -> A-layout via wave-private LDS tile, P@V via WMMA
//  - row-sum kept as per-lane partials, reduced once at the end
// ---------------------------------------------------------------------------
__global__ __launch_bounds__(128) void flash_kernel(
    const __bf16* __restrict__ FQ, const __bf16* __restrict__ FK,
    const __bf16* __restrict__ Vt, float* __restrict__ out) {
  __shared__ __align__(32) __bf16 pl[4][16][32];
  int lane = threadIdx.x & 31, w = threadIdx.x >> 5;
  int batch = blockIdx.y;
  int m0 = (blockIdx.x * 4 + w) * 16;
  int nl = lane & 15, hh = lane >> 4;
  const __bf16* Q = FQ + ((size_t)batch * SEQ + m0) * D_HID;
  const __bf16* K = FK + (size_t)batch * SEQ * D_HID;
  const __bf16* V = Vt + (size_t)batch * D_IN * SEQ;

  v16bf qf[8];
#pragma unroll
  for (int kk = 0; kk < 8; ++kk) qf[kk] = load_a_bf16(Q, D_HID, kk * 32, lane);

  v8f od[8];
#pragma unroll
  for (int dt = 0; dt < 8; ++dt) od[dt] = fzero();
  float rm[8], rsp[8];
#pragma unroll
  for (int v = 0; v < 8; ++v) { rm[v] = -3.0e38f; rsp[v] = 0.f; }

  for (int n0 = 0; n0 < SEQ; n0 += 32) {
    // e tile: 16 rows x 32 cols, two 16x16 WMMA groups
    const __bf16* kb0 = K + (size_t)n0 * D_HID;
    const __bf16* kb1 = kb0 + (size_t)16 * D_HID;
    __builtin_prefetch(kb0 + (size_t)32 * D_HID, 0, 0);
    v8f e0 = fzero(), e1 = fzero();
#pragma unroll
    for (int kk = 0; kk < 8; ++kk)
      e0 = wmma_bf16(qf[kk], load_b_bf16(kb0, D_HID, kk * 32, lane), e0);
#pragma unroll
    for (int kk = 0; kk < 8; ++kk)
      e1 = wmma_bf16(qf[kk], load_b_bf16(kb1, D_HID, kk * 32, lane), e1);

    // online softmax update (rescale factor uniform across each row's lanes)
    float sc[8];
#pragma unroll
    for (int v = 0; v < 8; ++v) {
      float tm = fmaxf(e0[v], e1[v]);
#pragma unroll
      for (int off = 1; off < 16; off <<= 1)
        tm = fmaxf(tm, __shfl_xor(tm, off, 32));
      float nm = fmaxf(rm[v], tm);
      sc[v] = __expf(rm[v] - nm);
      rm[v] = nm;
      float p0 = __expf(e0[v] - nm);
      float p1 = __expf(e1[v] - nm);
      rsp[v] = rsp[v] * sc[v] + p0 + p1;
      pl[w][v + hh * 8][nl] = (__bf16)p0;
      pl[w][v + hh * 8][16 + nl] = (__bf16)p1;
    }
#pragma unroll
    for (int dt = 0; dt < 8; ++dt)
#pragma unroll
      for (int v = 0; v < 8; ++v) od[dt][v] *= sc[v];

    // reload P as A-fragment (same-wave LDS, in-order), accumulate P@V
    v16bf pf = load_a_bf16(&pl[w][0][0], 32, 0, lane);
#pragma unroll
    for (int dt = 0; dt < 8; ++dt) {
      v16bf bf = load_b_bf16(V + (size_t)(dt * 16) * SEQ, SEQ, n0, lane);
      od[dt] = wmma_bf16(pf, bf, od[dt]);
    }
  }

  // finalize: reduce row-sum partials across 16 lanes, normalize, store
  float inv[8];
#pragma unroll
  for (int v = 0; v < 8; ++v) {
    float s = rsp[v];
#pragma unroll
    for (int off = 1; off < 16; off <<= 1) s += __shfl_xor(s, off, 32);
    inv[v] = 1.0f / s;
  }
#pragma unroll
  for (int dt = 0; dt < 8; ++dt)
#pragma unroll
    for (int v = 0; v < 8; ++v)
      out[((size_t)batch * SEQ + m0 + v + hh * 8) * D_IN + dt * 16 + nl] =
          od[dt][v] * inv[v];
}

// ---------------------------------------------------------------------------
extern "C" void kernel_launch(void* const* d_in, const int* in_sizes, int n_in,
                              void* d_out, int out_size, void* d_ws, size_t ws_size,
                              hipStream_t stream) {
  (void)in_sizes; (void)n_in; (void)out_size; (void)ws_size;
  const float* A  = (const float*)d_in[0];
  const float* B  = (const float*)d_in[1];
  const float* W1 = (const float*)d_in[2];
  const float* b1 = (const float*)d_in[3];
  const float* W2 = (const float*)d_in[4];
  const float* b2 = (const float*)d_in[5];
  float* beta  = (float*)d_out;
  float* alpha = beta + (size_t)NB * SEQ * D_IN;

  char* p = (char*)d_ws;
  __bf16* fA  = (__bf16*)p; p += (size_t)NB * SEQ * D_HID * 2;
  __bf16* fB  = (__bf16*)p; p += (size_t)NB * SEQ * D_HID * 2;
  __bf16* At  = (__bf16*)p; p += (size_t)NB * D_IN * SEQ * 2;
  __bf16* Bt  = (__bf16*)p; p += (size_t)NB * D_IN * SEQ * 2;
  __bf16* W1T = (__bf16*)p; p += (size_t)D_HID * D_IN * 2;
  __bf16* W2T = (__bf16*)p; p += (size_t)D_HID * D_HID * 2;

  prep_weights_kernel<<<64, 256, 0, stream>>>(W1, W2, W1T, W2T);
  transpose_kernel<<<2048, 256, 0, stream>>>(A, At);
  transpose_kernel<<<2048, 256, 0, stream>>>(B, Bt);
  // 16384 rows / 16 per wave / 4 waves per block = 256 blocks
  mlp_kernel<<<256, 128, 0, stream>>>(A, W1T, b1, W2T, b2, fA);
  mlp_kernel<<<256, 128, 0, stream>>>(B, W1T, b1, W2T, b2, fB);
  // beta  = rowsoftmax(fA·fBᵀ) @ B   (fused flash, single pass over n)
  // alpha = rowsoftmax(fB·fAᵀ) @ A   (same kernel, roles swapped)
  flash_kernel<<<dim3(64, NB), 128, 0, stream>>>(fA, fB, Bt, beta);
  flash_kernel<<<dim3(64, NB), 128, 0, stream>>>(fB, fA, At, alpha);
}
